// SingleStreamBlock_7576322310948
// MI455X (gfx1250) — matrix operand order repaired and checked by reading, not compiled, not run on hardware
//
#include <hip/hip_runtime.h>
#include <hip/hip_bf16.h>
#include <math.h>
#include <stdint.h>

#define HID 3072
#define HEADS 24
#define HD 128
#define MLP 12288
#define SEQ 2048
#define N1 (3*HID + MLP)   // 21504 = lin1 output dim
#define K2 (HID + MLP)     // 15360 = lin2 input dim

typedef __attribute__((ext_vector_type(16))) __bf16 v16bf;
typedef __attribute__((ext_vector_type(8)))  float  v8f;
typedef __attribute__((ext_vector_type(4)))  unsigned uint32x4;
typedef __attribute__((ext_vector_type(8)))  int      int32x8;

union FragAB { unsigned u[8]; v16bf v; };

static __device__ __forceinline__ unsigned short f2bf(float f) {
  unsigned u = __float_as_uint(f);
  u += 0x7FFFu + ((u >> 16) & 1u);           // round-to-nearest-even
  return (unsigned short)(u >> 16);
}
static __device__ __forceinline__ unsigned pk2(float a, float b) {
  return (unsigned)f2bf(a) | ((unsigned)f2bf(b) << 16);
}

// -------- CDNA5 async/TDM helpers (inline asm; ISA 15.18.3 / 7.x) ----------
static __device__ __forceinline__ unsigned lds_off32(const void* p) {
  // flat ptr to LDS keeps workgroup-relative offset in addr[31:0]
  return (unsigned)(uintptr_t)p;
}
// per-lane 16B memory->LDS copy, tracked with ASYNCcnt (GVS mode)
static __device__ __forceinline__ void async_load_b128(unsigned ldsAddr, unsigned gOff,
                                                       const void* base) {
  asm volatile("global_load_async_to_lds_b128 %0, %1, %2 offset:0"
               :: "v"(ldsAddr), "v"(gOff), "s"(base) : "memory");
}
static __device__ __forceinline__ void wait_asynccnt0() {
  asm volatile("s_wait_asynccnt 0x0" ::: "memory");
}
// TDM tile DMA: D# group0 (4 SGPRs) + group1 (8 SGPRs), 2-D tensor form
static __device__ __forceinline__ void tdm_load_2d(uint32x4 g0, int32x8 g1) {
  asm volatile("tensor_load_to_lds %0, %1" :: "s"(g0), "s"(g1) : "memory");
}

// ---------------------------------------------------------------------------
// 1) m = silu(vec) @ mod_w^T + mod_b        (9216 x 3072 GEMV, wave-per-row)
// ---------------------------------------------------------------------------
__global__ __launch_bounds__(256) void mod_gemv(const float* __restrict__ vec,
                                                const float* __restrict__ w,
                                                const float* __restrict__ b,
                                                float* __restrict__ m_out) {
  const int lane = threadIdx.x & 31, wave = threadIdx.x >> 5;
  const int row = blockIdx.x * 8 + wave;
  const float* wr = w + (size_t)row * HID;
  float acc = 0.f;
  for (int k = lane; k < HID; k += 32) {
    float v = vec[k];
    float sv = v / (1.f + __expf(-v));
    acc += sv * wr[k];
  }
  #pragma unroll
  for (int o = 16; o; o >>= 1) acc += __shfl_xor(acc, o, 32);
  if (lane == 0) m_out[row] = acc + b[row];
}

// ---------------------------------------------------------------------------
// 2) x_mod = (1+scale)*LN(x) + shift  -> bf16     (block per row)
// ---------------------------------------------------------------------------
__global__ __launch_bounds__(256) void ln_mod(const float* __restrict__ x,
                                              const float* __restrict__ m,
                                              unsigned short* __restrict__ xmod) {
  __shared__ float xs[HID];
  __shared__ float red[256];
  const int row = blockIdx.x, t = threadIdx.x;
  const float* xr = x + (size_t)row * HID;
  float s = 0.f;
  for (int n = t; n < HID; n += 256) { float v = xr[n]; xs[n] = v; s += v; }
  red[t] = s; __syncthreads();
  for (int o = 128; o; o >>= 1) { if (t < o) red[t] += red[t + o]; __syncthreads(); }
  const float mu = red[0] / (float)HID;
  __syncthreads();
  float s2 = 0.f;
  for (int n = t; n < HID; n += 256) { float d = xs[n] - mu; s2 += d * d; }
  red[t] = s2; __syncthreads();
  for (int o = 128; o; o >>= 1) { if (t < o) red[t] += red[t + o]; __syncthreads(); }
  const float rstd = rsqrtf(red[0] / (float)HID + 1e-6f);
  for (int n = t; n < HID; n += 256) {
    float ln = (xs[n] - mu) * rstd;
    float v = (1.f + m[HID + n]) * ln + m[n];
    xmod[(size_t)row * HID + n] = f2bf(v);
  }
}

// ---------------------------------------------------------------------------
// 3/6) bf16 WMMA GEMM: C[M,N] = A[M,K](bf16) @ B[N,K]^T(f32->bf16) + bias
//   A tile staged with GLOBAL_LOAD_ASYNC_TO_LDS_B128 (ASYNCcnt),
//   B tile converted f32->bf16 through VGPRs. 128x128 tile, KT=64,
//   double-buffered, 8 waves (2x4), v_wmma_f32_16x16x32_bf16.
// ---------------------------------------------------------------------------
template <int MODE>
__global__ __launch_bounds__(256) void gemm_wmma(
    const unsigned short* __restrict__ A, const float* __restrict__ Bw,
    const float* __restrict__ bias, int K,
    float* __restrict__ outF, unsigned short* __restrict__ outH,
    const float* __restrict__ xres, const float* __restrict__ gate) {
  __shared__ unsigned As[2][128 * 32];   // 128 rows x 64 bf16 (as 32 u32)
  __shared__ unsigned Bs[2][128 * 32];
  const int t = threadIdx.x;
  const int lane = t & 31, wave = t >> 5;
  const int wm = wave >> 2, wn = wave & 3;
  const int hl = lane >> 4, l16 = lane & 15;
  const int mBase = blockIdx.y * 128, nBase = blockIdx.x * 128;
  const int nt = K >> 6;
  const void* Abase = (const void*)(A + (size_t)mBase * K);   // uniform

  float4 bReg[8];
  auto loadA_async = [&](int kt) {                  // 128x64 bf16 tile, 16B/lane x4
    const int pb = kt & 1;
    #pragma unroll
    for (int i = 0; i < 4; ++i) {
      int s = t + i * 256, row = s >> 3, c4 = s & 7;
      unsigned gOff = (unsigned)((row * K + kt * 64 + c4 * 8) * 2);
      async_load_b128(lds_off32(&As[pb][row * 32 + c4 * 4]), gOff, Abase);
    }
  };
  auto loadB = [&](int kt) {                        // 128x64 f32 = 2048 float4
    #pragma unroll
    for (int i = 0; i < 8; ++i) {
      int s = t + i * 256, row = s >> 4, c4 = s & 15;
      bReg[i] = *(const float4*)(Bw + (size_t)(nBase + row) * K + kt * 64 + c4 * 4);
    }
    if (kt + 1 < nt)                                // -> global_prefetch_b8
      __builtin_prefetch(Bw + (size_t)(nBase + (t >> 1)) * K + (kt + 1) * 64 + (t & 1) * 32, 0, 1);
  };
  auto storeB = [&](int pb) {
    #pragma unroll
    for (int i = 0; i < 8; ++i) {
      int s = t + i * 256, row = s >> 4, c4 = s & 15;
      uint2 p; p.x = pk2(bReg[i].x, bReg[i].y); p.y = pk2(bReg[i].z, bReg[i].w);
      *(uint2*)&Bs[pb][row * 32 + c4 * 2] = p;
    }
  };

  v8f acc[4][2];
  #pragma unroll
  for (int f = 0; f < 4; ++f)
    #pragma unroll
    for (int g = 0; g < 2; ++g) acc[f][g] = {};

  loadA_async(0);
  loadB(0); storeB(0);
  for (int kt = 0; kt < nt; ++kt) {
    wait_asynccnt0();                               // my tile-kt A data landed
    __syncthreads();                                // everyone's tile visible
    if (kt + 1 < nt) { loadA_async(kt + 1); loadB(kt + 1); }
    const unsigned* Ab = As[kt & 1];
    const unsigned* Bb = Bs[kt & 1];
    #pragma unroll
    for (int ks = 0; ks < 2; ++ks) {                // two k=32 wmma steps per tile
      const int ko = ks * 16;
      FragAB af[4];
      #pragma unroll
      for (int f = 0; f < 4; ++f) {                 // A frag: lane=M, halves K0-7|16-23 / K8-15|24-31
        const unsigned* p = &Ab[(wm * 64 + f * 16 + l16) * 32 + ko];
        #pragma unroll
        for (int i = 0; i < 4; ++i) { af[f].u[i] = p[hl * 4 + i]; af[f].u[4 + i] = p[8 + hl * 4 + i]; }
      }
      FragAB bfr[2];
      #pragma unroll
      for (int g = 0; g < 2; ++g) {                 // B frag: lane=N, halves K0-7|16-23 / K8-15|24-31
        const unsigned* p = &Bb[(wn * 32 + g * 16 + l16) * 32 + ko];
        #pragma unroll
        for (int i = 0; i < 8; ++i) bfr[g].u[i] = p[(i >> 2) * 8 + hl * 4 + (i & 3)];
      }
      #pragma unroll
      for (int f = 0; f < 4; ++f)
        #pragma unroll
        for (int g = 0; g < 2; ++g)
          acc[f][g] = __builtin_amdgcn_wmma_f32_16x16x32_bf16(
              false, af[f].v, false, bfr[g].v, (short)0, acc[f][g], false, false);
    }
    if (kt + 1 < nt) storeB((kt + 1) & 1);
  }

  // epilogue: C layout -> row m = base + r + 8*hl, col n = base + l16
  #pragma unroll
  for (int f = 0; f < 4; ++f) {
    #pragma unroll
    for (int g = 0; g < 2; ++g) {
      const int n = nBase + wn * 32 + g * 16 + l16;
      const float bn = bias[n];
      #pragma unroll
      for (int r = 0; r < 8; ++r) {
        const int m = mBase + wm * 64 + f * 16 + r + 8 * hl;
        float v = acc[f][g][r] + bn;
        if (MODE == 1) {
          if (n < 3 * HID) {
            outF[(size_t)m * (3 * HID) + n] = v;
          } else {
            float ge = 0.5f * v * (1.f + tanhf(0.7978845608028654f * (v + 0.044715f * v * v * v)));
            outH[(size_t)m * K2 + HID + (n - 3 * HID)] = f2bf(ge);
          }
        } else {
          outF[(size_t)m * HID + n] = xres[(size_t)m * HID + n] + gate[n] * v;
        }
      }
    }
  }
}

// ---------------------------------------------------------------------------
// 4) RMSNorm + RoPE -> bf16 Q/K/V in [head][seq][128]   (wave per (m,head))
// ---------------------------------------------------------------------------
__global__ __launch_bounds__(256) void qkv_prep(
    const float* __restrict__ h, const float* __restrict__ pe,
    const float* __restrict__ qs, const float* __restrict__ ksc,
    unsigned short* __restrict__ qb, unsigned short* __restrict__ kb,
    unsigned short* __restrict__ vb) {
  const int lane = threadIdx.x & 31, wave = threadIdx.x >> 5;
  const int idx = blockIdx.x * 8 + wave;
  const int mrow = idx / HEADS, head = idx % HEADS;
  const float* qp = h + (size_t)mrow * (3 * HID) + head * HD;
  const float* kp = qp + HID;
  const float* vp = qp + 2 * HID;
  const int d0 = lane * 4;
  float4 q = *(const float4*)(qp + d0);
  float4 k = *(const float4*)(kp + d0);
  float4 v = *(const float4*)(vp + d0);
  float ssq = q.x * q.x + q.y * q.y + q.z * q.z + q.w * q.w;
  float ssk = k.x * k.x + k.y * k.y + k.z * k.z + k.w * k.w;
  #pragma unroll
  for (int o = 16; o; o >>= 1) { ssq += __shfl_xor(ssq, o, 32); ssk += __shfl_xor(ssk, o, 32); }
  const float rq = rsqrtf(ssq / (float)HD + 1e-6f);
  const float rk = rsqrtf(ssk / (float)HD + 1e-6f);
  float q0 = q.x * rq * qs[d0],     q1 = q.y * rq * qs[d0 + 1];
  float q2 = q.z * rq * qs[d0 + 2], q3 = q.w * rq * qs[d0 + 3];
  float k0 = k.x * rk * ksc[d0],     k1 = k.y * rk * ksc[d0 + 1];
  float k2 = k.z * rk * ksc[d0 + 2], k3 = k.w * rk * ksc[d0 + 3];
  const float* p0 = pe + (size_t)mrow * 256 + (size_t)lane * 8;  // [64][2][2]
  float e00 = p0[0], e01 = p0[1], e10 = p0[2], e11 = p0[3];
  float f00 = p0[4], f01 = p0[5], f10 = p0[6], f11 = p0[7];
  float oq0 = e00 * q0 + e01 * q1, oq1 = e10 * q0 + e11 * q1;
  float oq2 = f00 * q2 + f01 * q3, oq3 = f10 * q2 + f11 * q3;
  float ok0 = e00 * k0 + e01 * k1, ok1 = e10 * k0 + e11 * k1;
  float ok2 = f00 * k2 + f01 * k3, ok3 = f10 * k2 + f11 * k3;
  const size_t base = ((size_t)head * SEQ + mrow) * HD + d0;
  unsigned* qo = (unsigned*)(qb + base); qo[0] = pk2(oq0, oq1); qo[1] = pk2(oq2, oq3);
  unsigned* ko = (unsigned*)(kb + base); ko[0] = pk2(ok0, ok1); ko[1] = pk2(ok2, ok3);
  unsigned* vo = (unsigned*)(vb + base); vo[0] = pk2(v.x, v.y); vo[1] = pk2(v.z, v.w);
}

// ---------------------------------------------------------------------------
// 5) Flash attention: block = (head, 64 q-rows), 4 waves x 16 q-rows.
//    K tile staged by TDM TENSOR_LOAD_TO_LDS (TENSORcnt), V staged transposed,
//    S = Q@K^T and O += P@V via v_wmma_f32_16x16x32_bf16.
// ---------------------------------------------------------------------------
__global__ __launch_bounds__(128) void attn_fa(
    const unsigned short* __restrict__ qb, const unsigned short* __restrict__ kb,
    const unsigned short* __restrict__ vb, unsigned short* __restrict__ cat) {
  __shared__ unsigned       ksh[32][64];      // [key][d] bf16 pairs (TDM dest)
  __shared__ unsigned short vsh[128][32];     // [d][key] (transposed)
  __shared__ unsigned short psh[4][16][32];   // per-wave P bounce buffer
  const int head = blockIdx.x, qt = blockIdx.y;
  const int t = threadIdx.x, lane = t & 31, wave = t >> 5;
  const int hl = lane >> 4, l16 = lane & 15;
  const int qrow = qt * 64 + wave * 16;

  FragAB qf[4];                               // Q 16x128 = 4 A-frags of 16x32
  {
    const unsigned* qp = (const unsigned*)(qb + (size_t)(head * SEQ + qrow + l16) * HD);
    #pragma unroll
    for (int kk = 0; kk < 4; ++kk)
      #pragma unroll
      for (int i = 0; i < 4; ++i) {
        qf[kk].u[i]     = qp[kk * 16 + hl * 4 + i];
        qf[kk].u[4 + i] = qp[kk * 16 + 8 + hl * 4 + i];
      }
  }
  float mi[8], li[8];
  v8f o[8] = {};
  #pragma unroll
  for (int r = 0; r < 8; ++r) { mi[r] = -1e30f; li[r] = 0.f; }
  const float sc = 0.08838834764831845f;      // 1/sqrt(128)

  for (int kt = 0; kt < SEQ; kt += 32) {
    __syncthreads();
    #pragma unroll
    for (int i = 0; i < 4; ++i) {             // stage V transposed (all threads)
      int s = t + i * 128, key = s >> 4, c4 = s & 15;
      uint4 vd = *(const uint4*)(vb + (size_t)(head * SEQ + kt + key) * HD + c4 * 8);
      const unsigned short* pv = (const unsigned short*)&vd;
      #pragma unroll
      for (int e = 0; e < 8; ++e) vsh[c4 * 8 + e][key] = pv[e];
    }
    if (wave == 0) {                          // TDM: 32 rows x 256B K tile -> ksh
      unsigned long long ga =
          (unsigned long long)(uintptr_t)(kb + (size_t)(head * SEQ + kt) * HD);
      uint32x4 g0;
      g0.x = 1u;                                   // count=1, user mode
      g0.y = lds_off32(&ksh[0][0]);                // lds_addr
      g0.z = (unsigned)ga;                         // global_addr[31:0]
      g0.w = (unsigned)((ga >> 32) & 0x1FFFFFFu) | (2u << 30);  // addr[56:32] | type=2
      int32x8 g1;
      g1[0] = 0x00030000;                          // wg_mask=0, data_size=8B
      g1[1] = 32 << 16;                            // tensor_dim0 = 32 units (256B row)
      g1[2] = (int)(2048u << 16);                  // tensor_dim1[15:0] = 2048 rows
      g1[3] = 32 << 16;                            // tile_dim0 = 32 units
      g1[4] = 32;                                  // tile_dim1 = 32 rows
      g1[5] = 32;                                  // tensor_dim0_stride = 32 units
      g1[6] = 0;
      g1[7] = 0;
      tdm_load_2d(g0, g1);
      __builtin_amdgcn_s_wait_tensorcnt(0);
    }
    __syncthreads();

    v8f s0 = {}, s1 = {};
    #pragma unroll
    for (int kk = 0; kk < 4; ++kk) {          // S(16x32 keys) over d=128
      FragAB b0, b1;
      #pragma unroll
      for (int i = 0; i < 8; ++i) {
        b0.u[i] = ksh[l16][kk * 16 + (i >> 2) * 8 + hl * 4 + (i & 3)];
        b1.u[i] = ksh[16 + l16][kk * 16 + (i >> 2) * 8 + hl * 4 + (i & 3)];
      }
      s0 = __builtin_amdgcn_wmma_f32_16x16x32_bf16(false, qf[kk].v, false, b0.v, (short)0, s0, false, false);
      s1 = __builtin_amdgcn_wmma_f32_16x16x32_bf16(false, qf[kk].v, false, b1.v, (short)0, s1, false, false);
    }
    #pragma unroll
    for (int r = 0; r < 8; ++r) {             // online softmax (row = r + 8*hl)
      float a0 = s0[r] * sc, a1 = s1[r] * sc;
      float rm = fmaxf(a0, a1);
      #pragma unroll
      for (int d = 8; d >= 1; d >>= 1) rm = fmaxf(rm, __shfl_xor(rm, d, 32));
      float mnew = fmaxf(mi[r], rm);
      float alpha = __expf(mi[r] - mnew);
      float p0 = __expf(a0 - mnew), p1 = __expf(a1 - mnew);
      float rs = p0 + p1;
      #pragma unroll
      for (int d = 8; d >= 1; d >>= 1) rs += __shfl_xor(rs, d, 32);
      li[r] = li[r] * alpha + rs;
      mi[r] = mnew;
      #pragma unroll
      for (int g = 0; g < 8; ++g) o[g][r] *= alpha;
      psh[wave][r + 8 * hl][l16]      = f2bf(p0);
      psh[wave][r + 8 * hl][16 + l16] = f2bf(p1);
    }
    FragAB pa;                                // C-layout -> A-layout via LDS bounce
    {
      const unsigned* pp = (const unsigned*)&psh[wave][l16][0];
      #pragma unroll
      for (int i = 0; i < 4; ++i) { pa.u[i] = pp[hl * 4 + i]; pa.u[4 + i] = pp[8 + hl * 4 + i]; }
    }
    #pragma unroll
    for (int g = 0; g < 8; ++g) {             // O(16x128) += P(16x32) @ V(32x128)
      FragAB bv;
      #pragma unroll
      for (int i = 0; i < 8; ++i)
        bv.u[i] = *(const unsigned*)&vsh[g * 16 + l16][(i >> 2) * 16 + hl * 8 + 2 * (i & 3)];
      o[g] = __builtin_amdgcn_wmma_f32_16x16x32_bf16(false, pa.v, false, bv.v, (short)0, o[g], false, false);
    }
  }
  #pragma unroll
  for (int r = 0; r < 8; ++r) {
    const float inv = 1.f / li[r];
    const int m = qrow + r + 8 * hl;
    #pragma unroll
    for (int g = 0; g < 8; ++g)
      cat[(size_t)m * K2 + head * HD + g * 16 + l16] = f2bf(o[g][r] * inv);
  }
}

// ---------------------------------------------------------------------------
extern "C" void kernel_launch(void* const* d_in, const int* in_sizes, int n_in,
                              void* d_out, int out_size, void* d_ws, size_t ws_size,
                              hipStream_t stream) {
  (void)in_sizes; (void)n_in; (void)out_size; (void)ws_size;
  const float* x       = (const float*)d_in[0];
  const float* vec     = (const float*)d_in[1];
  const float* pe      = (const float*)d_in[2];
  const float* mod_w   = (const float*)d_in[3];
  const float* mod_b   = (const float*)d_in[4];
  const float* lin1_w  = (const float*)d_in[5];
  const float* lin1_b  = (const float*)d_in[6];
  const float* lin2_w  = (const float*)d_in[7];
  const float* lin2_b  = (const float*)d_in[8];
  const float* q_scale = (const float*)d_in[9];
  const float* k_scale = (const float*)d_in[10];
  float* out = (float*)d_out;

  char* ws = (char*)d_ws;
  size_t off = 0;
  auto alloc = [&](size_t bytes) -> void* {
    void* p = ws + off;
    off += (bytes + 255) & ~(size_t)255;
    return p;
  };
  float*          m_out = (float*)alloc((size_t)3 * HID * 4);
  unsigned short* xmod  = (unsigned short*)alloc((size_t)SEQ * HID * 2);
  float*          hqkv  = (float*)alloc((size_t)SEQ * 3 * HID * 4);
  unsigned short* catb  = (unsigned short*)alloc((size_t)SEQ * K2 * 2);
  unsigned short* qb    = (unsigned short*)alloc((size_t)HEADS * SEQ * HD * 2);
  unsigned short* kb    = (unsigned short*)alloc((size_t)HEADS * SEQ * HD * 2);
  unsigned short* vb    = (unsigned short*)alloc((size_t)HEADS * SEQ * HD * 2);

  mod_gemv<<<(3 * HID) / 8, 256, 0, stream>>>(vec, mod_w, mod_b, m_out);
  ln_mod<<<SEQ, 256, 0, stream>>>(x, m_out, xmod);
  gemm_wmma<1><<<dim3(N1 / 128, SEQ / 128), 256, 0, stream>>>(
      xmod, lin1_w, lin1_b, HID, hqkv, catb, nullptr, nullptr);
  qkv_prep<<<(SEQ * HEADS) / 8, 256, 0, stream>>>(hqkv, pe, q_scale, k_scale, qb, kb, vb);
  attn_fa<<<dim3(HEADS, SEQ / 64), 128, 0, stream>>>(qb, kb, vb, catb);
  gemm_wmma<2><<<dim3(HID / 128, SEQ / 128), 256, 0, stream>>>(
      catb, lin2_w, lin2_b, K2, out, nullptr, x, m_out + 2 * HID);
}
